// ImprovedRPPGModel_44392781971769
// MI455X (gfx1250) — compile-verified, matrix operand order checked by reading, not run on hardware
//
#include <hip/hip_runtime.h>
#include <hip/hip_bf16.h>
#include <math.h>

// ---------------- model dims ----------------
#define Bb 4
#define Ll 2048
#define DM 128
#define DS 16
#define NB 6
#define DI 256          // D_INNER
#define DTR 8           // DT_RANK
#define DTRP 32         // DT_RANK padded to one WMMA K-step
#define CK 7            // pre conv kernel
#define DC 8            // mamba depthwise conv kernel
#define MM (Bb*Ll)      // 8192 rows
#define NXP 48          // W_x N=40 padded to 3 tiles

typedef __attribute__((ext_vector_type(16))) __bf16 v16bf;
typedef __attribute__((ext_vector_type(8)))  __bf16 v8bf;
typedef __attribute__((ext_vector_type(8)))  float  v8f;

__device__ __forceinline__ __bf16 f2bf(float x) { return (__bf16)x; }
__device__ __forceinline__ float  bf2f(__bf16 x) { return (float)x; }
__device__ __forceinline__ float  silu_f(float v) { return v / (1.0f + __expf(-v)); }
__device__ __forceinline__ v16bf  cat8(v8bf lo, v8bf hi) {
    return __builtin_shufflevector(lo, hi, 0,1,2,3,4,5,6,7,8,9,10,11,12,13,14,15);
}

// ---------------- f32 -> bf16 convert ----------------
__global__ void k_cvt_bf16(const float* __restrict__ src, __bf16* __restrict__ dst, int n) {
    int i = blockIdx.x * blockDim.x + threadIdx.x;
    if (i < n) dst[i] = f2bf(src[i]);
}

// f32 -> bf16 with K padding: dst (rows, Kdst), src rows stride srcStride, valid K = Ksrc
__global__ void k_cvt_pad(const float* __restrict__ src, __bf16* __restrict__ dst,
                          int rows, int srcStride, int Ksrc, int Kdst) {
    int i = blockIdx.x * blockDim.x + threadIdx.x;
    if (i >= rows * Kdst) return;
    int r = i / Kdst, k = i - r * Kdst;
    dst[i] = (k < Ksrc) ? f2bf(src[(size_t)r * srcStride + k]) : f2bf(0.f);
}

// W_x: (NB,40,256) f32 -> (NB,48,256) bf16, rows 40..47 zero
__global__ void k_cvt_wx(const float* __restrict__ src, __bf16* __restrict__ dst) {
    int i = blockIdx.x * blockDim.x + threadIdx.x;
    if (i >= NB * NXP * DI) return;
    int blk = i / (NXP * DI);
    int rem = i - blk * (NXP * DI);
    int r = rem / DI, k = rem - r * DI;
    dst[i] = (r < DTR + 2 * DS)
           ? f2bf(src[((size_t)blk * (DTR + 2 * DS) + r) * DI + k]) : f2bf(0.f);
}

// ---------------- pre conv + exact GELU -> hbuf (M,128) f32 ----------------
__global__ void k_preconv(const float* __restrict__ x, const float* __restrict__ pw,
                          const float* __restrict__ pb, float* __restrict__ hbuf) {
    int idx = blockIdx.x * blockDim.x + threadIdx.x;
    if (idx >= MM * DM) return;
    int d = idx & (DM - 1);
    int m = idx >> 7;
    int b = m / Ll, l = m - b * Ll;
    float acc = pb[d];
#pragma unroll
    for (int k = 0; k < CK; ++k) {
        int li = l + k - CK / 2;
        if (li >= 0 && li < Ll) acc += x[b * Ll + li] * pw[d * CK + k];
    }
    hbuf[m * DM + d] = 0.5f * acc * (1.0f + erff(acc * 0.70710678118654752f));
}

// ---------------- layernorm row -> bf16 (M,128) ----------------
__global__ void k_ln_bf16(const float* __restrict__ hbuf,
                          const float* __restrict__ lw, const float* __restrict__ lb,
                          __bf16* __restrict__ out) {
    int m = blockIdx.x * blockDim.x + threadIdx.x;
    if (m >= MM) return;
    const float* r = hbuf + (size_t)m * DM;
    float mu = 0.f;
#pragma unroll 4
    for (int d = 0; d < DM; ++d) mu += r[d];
    mu *= (1.0f / DM);
    float var = 0.f;
#pragma unroll 4
    for (int d = 0; d < DM; ++d) { float t = r[d] - mu; var += t * t; }
    var *= (1.0f / DM);
    float rs = rsqrtf(var + 1e-5f);
    __bf16* o = out + (size_t)m * DM;
#pragma unroll 4
    for (int d = 0; d < DM; ++d) o[d] = f2bf((r[d] - mu) * rs * lw[d] + lb[d]);
}

// ---------------- bf16 WMMA GEMM: C[M,Nreal] = A[M,K] * W[Npad,K]^T ----------------
// Preconditions: M%16==0, Npad%(16*NT)==0, K%32==0, lda/ldw multiples of 8,
// buffers 16B aligned. Each wave computes NT adjacent N-tiles sharing one A frag.
// MODE 0: store; MODE 1: softplus(acc + bias[n]); MODE 2: C += acc
template <int MODE, int NT>
__global__ void k_gemm_wmma(const __bf16* __restrict__ A, int lda,
                            const __bf16* __restrict__ W, int ldw,
                            float* __restrict__ C, int ldc,
                            int M, int Npad, int Nreal, int K,
                            const float* __restrict__ bias) {
    const int wave = threadIdx.x >> 5;          // 4 waves / block
    const int lane = threadIdx.x & 31;
    const int ntn  = Npad / (16 * NT);          // N-tile groups
    int tile = blockIdx.x * 4 + wave;
    if (tile >= (M >> 4) * ntn) return;
    int mt = tile / ntn;
    int nt = tile - mt * ntn;

    const int kh = lane >> 4;                   // lane half -> K-group select
    const int mg = mt * 16 + (lane & 15);       // A row
    const int ng = nt * (16 * NT) + (lane & 15);// first W row (output col)

    const v8bf* __restrict__ Ar = (const v8bf*)(A + (size_t)mg * lda);
    const v8bf* __restrict__ Wr[NT];
#pragma unroll
    for (int t = 0; t < NT; ++t)
        Wr[t] = (const v8bf*)(W + (size_t)(ng + 16 * t) * ldw);

    v8f acc[NT];
#pragma unroll
    for (int t = 0; t < NT; ++t) acc[t] = (v8f){};

    const int ksteps = K >> 5;
    for (int ks = 0; ks < ksteps; ++ks) {
        int v0 = ks * 4 + kh;                   // K-groups {0..7 | 8..15}
        int v1 = ks * 4 + 2 + kh;               // K-groups {16..23 | 24..31}
        v16bf a = cat8(Ar[v0], Ar[v1]);
#pragma unroll
        for (int t = 0; t < NT; ++t) {
            v16bf b = cat8(Wr[t][v0], Wr[t][v1]);
            acc[t] = __builtin_amdgcn_wmma_f32_16x16x32_bf16(false, a, false, b,
                                                             (short)0, acc[t], false, false);
        }
    }

#pragma unroll
    for (int t = 0; t < NT; ++t) {
        int n = nt * (16 * NT) + 16 * t + (lane & 15);
        if (n >= Nreal) continue;
#pragma unroll
        for (int r = 0; r < 8; ++r) {
            int mmr = mt * 16 + r + 8 * kh;
            float v = acc[t][r];
            if (MODE == 1) {
                float u = v + bias[n];
                v = (u > 20.f) ? u : log1pf(__expf(u));   // softplus
            }
            if (MODE == 2) C[(size_t)mmr * ldc + n] += v;
            else           C[(size_t)mmr * ldc + n]  = v;
        }
    }
}

// ---------------- depthwise causal conv (k=8) + silu -> bf16 (M,256) ----------------
__global__ void k_dwconv(const float* __restrict__ xz, const float* __restrict__ cw,
                         const float* __restrict__ cb, __bf16* __restrict__ xcb) {
    int idx = blockIdx.x * blockDim.x + threadIdx.x;
    if (idx >= MM * DI) return;
    int e = idx & (DI - 1);
    int m = idx >> 8;
    int b = m / Ll, l = m - b * Ll;
    float acc = cb[e];
#pragma unroll
    for (int k = 0; k < DC; ++k) {
        int li = l - (DC - 1) + k;
        if (li >= 0) acc += xz[(size_t)(b * Ll + li) * (2 * DI) + e] * cw[e * DC + k];
    }
    xcb[(size_t)m * DI + e] = f2bf(silu_f(acc));
}

// ---------------- selective scan: one thread per (b,e), 16 states in regs ----------------
__global__ void k_scan(const float* __restrict__ delta, const __bf16* __restrict__ xcb,
                       const float* __restrict__ dbl, const float* __restrict__ Alog,
                       float* __restrict__ ys) {
    int t = blockIdx.x * blockDim.x + threadIdx.x;
    if (t >= Bb * DI) return;
    int b = t >> 8, e = t & (DI - 1);
    float Arow[DS];
#pragma unroll
    for (int n = 0; n < DS; ++n) Arow[n] = -__expf(Alog[e * DS + n]);
    float h[DS];
#pragma unroll
    for (int n = 0; n < DS; ++n) h[n] = 0.f;
    for (int l = 0; l < Ll; ++l) {
        size_t m = (size_t)b * Ll + l;
        float d  = delta[m * DI + e];
        float u  = bf2f(xcb[m * DI + e]);
        float du = d * u;
        const float* bc = dbl + m * (DTR + 2 * DS);
        float y = 0.f;
#pragma unroll
        for (int n = 0; n < DS; ++n) {
            h[n] = h[n] * __expf(d * Arow[n]) + du * bc[DTR + n];
            y += h[n] * bc[DTR + DS + n];
        }
        ys[m * DI + e] = y;
    }
}

// ---------------- gating: y = (ys + Dsk*xc) * silu(z) -> bf16 (M,256) ----------------
__global__ void k_gate(const float* __restrict__ ys, const __bf16* __restrict__ xcb,
                       const float* __restrict__ xz, const float* __restrict__ Dsk,
                       __bf16* __restrict__ yb) {
    int idx = blockIdx.x * blockDim.x + threadIdx.x;
    if (idx >= MM * DI) return;
    int e = idx & (DI - 1);
    int m = idx >> 8;
    float z = xz[(size_t)m * (2 * DI) + DI + e];
    float v = (ys[(size_t)m * DI + e] + Dsk[e] * bf2f(xcb[(size_t)m * DI + e])) * silu_f(z);
    yb[(size_t)m * DI + e] = f2bf(v);
}

// ---------------- final projection ----------------
__global__ void k_final(const float* __restrict__ hbuf,
                        const float* __restrict__ ow, const float* __restrict__ ob,
                        float* __restrict__ out) {
    int m = blockIdx.x * blockDim.x + threadIdx.x;
    if (m >= MM) return;
    const float* r = hbuf + (size_t)m * DM;
    float acc = ob[0];
#pragma unroll 4
    for (int d = 0; d < DM; ++d) acc += r[d] * ow[d];
    out[m] = acc;
}

// ============================ host side ============================
extern "C" void kernel_launch(void* const* d_in, const int* in_sizes, int n_in,
                              void* d_out, int out_size, void* d_ws, size_t ws_size,
                              hipStream_t stream) {
    const float* x_in   = (const float*)d_in[0];
    const float* pcw    = (const float*)d_in[1];
    const float* pcb    = (const float*)d_in[2];
    const float* ln_w   = (const float*)d_in[3];
    const float* ln_b   = (const float*)d_in[4];
    const float* W_in   = (const float*)d_in[5];   // (6,512,128)
    const float* conv_w = (const float*)d_in[6];   // (6,256,1,8)
    const float* conv_b = (const float*)d_in[7];
    const float* W_x    = (const float*)d_in[8];   // (6,40,256)
    const float* W_dt   = (const float*)d_in[9];   // (6,256,8)
    const float* b_dt   = (const float*)d_in[10];  // (6,256)
    const float* A_log  = (const float*)d_in[11];  // (6,256,16)
    const float* D_skip = (const float*)d_in[12];  // (6,256)
    const float* W_out  = (const float*)d_in[13];  // (6,128,256)
    const float* out_w  = (const float*)d_in[14];  // (1,128)
    const float* out_b  = (const float*)d_in[15];
    float* out = (float*)d_out;

    char* base = (char*)d_ws;
    size_t off = 0;
    auto carve = [&](size_t bytes) -> char* {
        char* p = base + off;
        off = (off + bytes + 255) & ~(size_t)255;
        return p;
    };
    float*  hbuf  = (float*) carve((size_t)MM * DM * 4);
    __bf16* lnb   = (__bf16*)carve((size_t)MM * DM * 2);
    float*  xz    = (float*) carve((size_t)MM * 2 * DI * 4);
    __bf16* xcb   = (__bf16*)carve((size_t)MM * DI * 2);
    float*  dbl   = (float*) carve((size_t)MM * (DTR + 2 * DS) * 4);
    __bf16* dtb   = (__bf16*)carve((size_t)MM * DTRP * 2);      // dt padded K=32
    float*  delta = (float*) carve((size_t)MM * DI * 4);
    float*  ys    = (float*) carve((size_t)MM * DI * 4);
    __bf16* yb    = (__bf16*)carve((size_t)MM * DI * 2);
    __bf16* winb  = (__bf16*)carve((size_t)NB * 2 * DI * DM * 2);
    __bf16* wxb   = (__bf16*)carve((size_t)NB * NXP * DI * 2);  // N padded to 48
    __bf16* wdtb  = (__bf16*)carve((size_t)NB * DI * DTRP * 2); // K padded to 32
    __bf16* woutb = (__bf16*)carve((size_t)NB * DM * DI * 2);

    // ---- weights -> bf16 (with padding) ----
    {
        int n1 = NB * 2 * DI * DM;
        k_cvt_bf16<<<(n1 + 255) / 256, 256, 0, stream>>>(W_in, winb, n1);
        int n2 = NB * NXP * DI;
        k_cvt_wx<<<(n2 + 255) / 256, 256, 0, stream>>>(W_x, wxb);
        int n3 = NB * DI * DTRP;
        k_cvt_pad<<<(n3 + 255) / 256, 256, 0, stream>>>(W_dt, wdtb, NB * DI, DTR, DTR, DTRP);
        int n4 = NB * DM * DI;
        k_cvt_bf16<<<(n4 + 255) / 256, 256, 0, stream>>>(W_out, woutb, n4);
    }

    // ---- stem ----
    k_preconv<<<(MM * DM + 255) / 256, 256, 0, stream>>>(x_in, pcw, pcb, hbuf);

    // ---- blocks ----
    for (int blk = 0; blk < NB; ++blk) {
        const __bf16* Wi  = winb  + (size_t)blk * 2 * DI * DM;
        const __bf16* Wx  = wxb   + (size_t)blk * NXP * DI;
        const __bf16* Wdt = wdtb  + (size_t)blk * DI * DTRP;
        const __bf16* Wo  = woutb + (size_t)blk * DM * DI;

        k_ln_bf16<<<(MM + 127) / 128, 128, 0, stream>>>(
            hbuf, ln_w + blk * DM, ln_b + blk * DM, lnb);

        // xz = lnb(8192x128) @ W_in(512x128)^T       (NT=2: 16 N-tile pairs)
        {
            int tiles = (MM / 16) * ((2 * DI) / 32);
            k_gemm_wmma<0, 2><<<(tiles + 3) / 4, 128, 0, stream>>>(
                lnb, DM, Wi, DM, xz, 2 * DI, MM, 2 * DI, 2 * DI, DM, nullptr);
        }

        k_dwconv<<<(MM * DI + 255) / 256, 256, 0, stream>>>(
            xz, conv_w + (size_t)blk * DI * DC, conv_b + blk * DI, xcb);

        // dbl = xcb(8192x256) @ W_x(48x256)^T, store N=40   (NT=1: 3 N-tiles)
        {
            int tiles = (MM / 16) * (NXP / 16);
            k_gemm_wmma<0, 1><<<(tiles + 3) / 4, 128, 0, stream>>>(
                xcb, DI, Wx, DI, dbl, DTR + 2 * DS, MM, NXP, DTR + 2 * DS, DI, nullptr);
        }

        // dt padded -> bf16 (M,32) from dbl rows (stride 40, first 8 valid)
        {
            int n = MM * DTRP;
            k_cvt_pad<<<(n + 255) / 256, 256, 0, stream>>>(
                dbl, dtb, MM, DTR + 2 * DS, DTR, DTRP);
        }

        // delta = softplus(dtb(8192x32) @ W_dt(256x32)^T + b_dt)   (NT=2)
        {
            int tiles = (MM / 16) * (DI / 32);
            k_gemm_wmma<1, 2><<<(tiles + 3) / 4, 128, 0, stream>>>(
                dtb, DTRP, Wdt, DTRP, delta, DI, MM, DI, DI, DTRP, b_dt + blk * DI);
        }

        k_scan<<<(Bb * DI + 255) / 256, 256, 0, stream>>>(
            delta, xcb, dbl, A_log + (size_t)blk * DI * DS, ys);

        k_gate<<<(MM * DI + 255) / 256, 256, 0, stream>>>(
            ys, xcb, xz, D_skip + blk * DI, yb);

        // hbuf += yb(8192x256) @ W_out(128x256)^T    (NT=2)
        {
            int tiles = (MM / 16) * (DM / 32);
            k_gemm_wmma<2, 2><<<(tiles + 3) / 4, 128, 0, stream>>>(
                yb, DI, Wo, DI, hbuf, DM, MM, DM, DM, DI, nullptr);
        }
    }

    k_final<<<(MM + 255) / 256, 256, 0, stream>>>(hbuf, out_w, out_b, out);

    (void)in_sizes; (void)n_in; (void)out_size; (void)ws_size;
}
